// WarpNet_64080912056669
// MI455X (gfx1250) — compile-verified, matrix operand order checked by reading, not run on hardware
//
#include <hip/hip_runtime.h>

// ---------------------------------------------------------------------------
// Types
// ---------------------------------------------------------------------------
typedef __bf16    bf16_t;
typedef _Float16  f16_t;
typedef __bf16    v16bf __attribute__((ext_vector_type(16)));
typedef _Float16  v16h  __attribute__((ext_vector_type(16)));
typedef float     v8f   __attribute__((ext_vector_type(8)));
typedef unsigned int u32x4 __attribute__((ext_vector_type(4)));
typedef unsigned int v4u  __attribute__((ext_vector_type(4)));
typedef int          v4i  __attribute__((ext_vector_type(4)));
typedef int          v8i  __attribute__((ext_vector_type(8)));

#ifndef __has_builtin
#define __has_builtin(x) 0
#endif
#if __has_builtin(__builtin_amdgcn_tensor_load_to_lds) && \
    __has_builtin(__builtin_amdgcn_s_wait_tensorcnt)
#define USE_TDM 1
#else
#define USE_TDM 0
#endif

template <typename ET> struct VecOf;
template <> struct VecOf<bf16_t> { using T = v16bf; };
template <> struct VecOf<f16_t>  { using T = v16h;  };

template <typename V> union FragU { u32x4 u[2]; V v; };

__device__ __forceinline__ v8f wmma_mma(v16bf a, v16bf b, v8f c) {
    return __builtin_amdgcn_wmma_f32_16x16x32_bf16(false, a, false, b, (short)0, c, false, false);
}
__device__ __forceinline__ v8f wmma_mma(v16h a, v16h b, v8f c) {
    return __builtin_amdgcn_wmma_f32_16x16x32_f16(false, a, false, b, (short)0, c, false, false);
}

#if USE_TDM
// ---------------------------------------------------------------------------
// Tensor Data Mover: 2D tile load (16-bit elements) global -> LDS.
// D# per CDNA5 ISA 8.3/8.4: group0 = {count=1 | lds_addr | global_addr | type=2},
// group1 = {data_size=2B, tensor_dim0/1, tile_dim0/1, tensor_dim0_stride}.
// Rows beyond tensor_dim1 read as zero (free ragged-M/N zero fill).
// ---------------------------------------------------------------------------
__device__ __forceinline__ void tdm_load_2d(
    unsigned lds_addr, const void* gptr,
    unsigned tensor_d0, unsigned tensor_d1,
    unsigned tile_d0, unsigned tile_d1,
    unsigned long long row_stride_elems)
{
    const unsigned long long ga = (unsigned long long)gptr;
    v4u g0;
    g0[0] = 1u;                                         // count=1, is_restore=0
    g0[1] = lds_addr;                                   // LDS byte address
    g0[2] = (unsigned)(ga & 0xFFFFFFFFu);               // global_addr[31:0]
    g0[3] = (unsigned)((ga >> 32) & 0x01FFFFFFu)        // global_addr[56:32]
            | 0x80000000u;                              // type = 2 ("image")
    v8i g1;
    g1[0] = 0x00010000;                                 // data_size = 1 -> 2 bytes
    g1[1] = (int)((tensor_d0 & 0xFFFFu) << 16);         // tensor_dim0[15:0]
    g1[2] = (int)((tensor_d0 >> 16) | ((tensor_d1 & 0xFFFFu) << 16));
    g1[3] = (int)((tensor_d1 >> 16) | ((tile_d0 & 0xFFFFu) << 16));
    g1[4] = (int)(tile_d1 & 0xFFFFu);                   // tile_dim1 (tile_dim2=0)
    g1[5] = (int)(row_stride_elems & 0xFFFFFFFFull);    // tensor_dim0_stride lo
    g1[6] = (int)((row_stride_elems >> 32) & 0xFFFFull);// stride hi (dim1_stride=0)
    g1[7] = 0;
    const v4i z4 = {0, 0, 0, 0};
#if __clang_major__ >= 23
    const v8i z8 = {0, 0, 0, 0, 0, 0, 0, 0};
    __builtin_amdgcn_tensor_load_to_lds(g0, g1, z4, z4, z8, 0);
#else
    __builtin_amdgcn_tensor_load_to_lds(g0, g1, z4, z4, 0);
#endif
}
#endif // USE_TDM

// ---------------------------------------------------------------------------
// NT-GEMM:  C[M,N] (f32) = A[M,K] * B[N,K]^T, A/B row-major over K (bf16|f16)
// 256 thr = 8 waves, block tile 128(M) x 128(N), K in 32-slabs via LDS.
// TDM path: wave 0 DMAs tiles (double-buffered, TENSORcnt-tracked); all waves
// only compute. Fallback: cooperative global->LDS staging.
// OT=true stores C transposed: C[n*ldc + m].
// ---------------------------------------------------------------------------
template <typename ET, bool OT>
__global__ __launch_bounds__(256) void gemm_nt(
    const ET* __restrict__ A, const ET* __restrict__ B, float* __restrict__ C,
    int M, int N, int K, int lda, int ldb, int ldc)
{
    using V = typename VecOf<ET>::T;
#if USE_TDM
    __shared__ alignas(16) ET As[2][128 * 32];
    __shared__ alignas(16) ET Bs[2][128 * 32];
#else
    __shared__ alignas(16) ET As[1][128 * 32];
    __shared__ alignas(16) ET Bs[1][128 * 32];
#endif
    const int t    = threadIdx.x;
    const int gm0  = blockIdx.y * 128;
    const int gn0  = blockIdx.x * 128;
    const int lane = t & 31;
    const int wave = t >> 5;
    const int wm   = (wave & 1) * 64;
    const int wn   = (wave >> 1) * 32;
    const int lm   = lane & 15;
    const int lk8  = (lane >> 4) * 8;   // per-lane K base inside each 16-wide half

    v8f acc[4][2];
    const v8f vzero = {0.f, 0.f, 0.f, 0.f, 0.f, 0.f, 0.f, 0.f};
#pragma unroll
    for (int i = 0; i < 4; ++i)
#pragma unroll
        for (int j = 0; j < 2; ++j) acc[i][j] = vzero;

#if USE_TDM
    const unsigned mrem = (unsigned)(M - gm0);
    const unsigned nrem = (unsigned)(N - gn0);
    const ET* atile = A + (size_t)gm0 * lda;
    const ET* btile = B + (size_t)gn0 * ldb;
    const unsigned ldsA = (unsigned)(unsigned long long)(const void*)&As[0][0];
    const unsigned ldsB = (unsigned)(unsigned long long)(const void*)&Bs[0][0];
    if (wave == 0) {
        tdm_load_2d(ldsA, atile, 32u, mrem, 32u, 128u, (unsigned long long)lda);
        tdm_load_2d(ldsB, btile, 32u, nrem, 32u, 128u, (unsigned long long)ldb);
        __builtin_amdgcn_s_wait_tensorcnt(0);
    }
    __syncthreads();
#else
    const int lrow  = t >> 1;            // 0..127 : tile row staged by this thread
    const int lhalf = (t & 1) * 16;      // 0 or 16 : K-half staged by this thread
    const int ar = (gm0 + lrow < M) ? gm0 + lrow : M - 1;   // clamp: OOB rows are
    const int br = (gn0 + lrow < N) ? gn0 + lrow : N - 1;   // garbage, never stored
    const ET* ap = A + (size_t)ar * lda + lhalf;
    const ET* bp = B + (size_t)br * ldb + lhalf;
#endif

    for (int k0 = 0; k0 < K; k0 += 32) {
#if USE_TDM
        const int cur = (k0 >> 5) & 1;
        if (wave == 0 && k0 + 32 < K) {   // prefetch next slab into other buffer
            const unsigned o = (unsigned)(cur ? 0 : 128 * 32 * sizeof(ET));
            tdm_load_2d(ldsA + o, atile + k0 + 32, 32u, mrem, 32u, 128u,
                        (unsigned long long)lda);
            tdm_load_2d(ldsB + o, btile + k0 + 32, 32u, nrem, 32u, 128u,
                        (unsigned long long)ldb);
        }
        const ET* Asc = &As[cur][0];
        const ET* Bsc = &Bs[cur][0];
#else
        const u32x4* pa = (const u32x4*)(ap + k0);
        const u32x4* pb = (const u32x4*)(bp + k0);
        u32x4 a0 = pa[0], a1 = pa[1], b0 = pb[0], b1 = pb[1];
        if (k0 + 32 < K) {                 // global_prefetch_b8 of next K slab
            __builtin_prefetch(ap + k0 + 32, 0, 1);
            __builtin_prefetch(bp + k0 + 32, 0, 1);
        }
        __syncthreads();
        *(u32x4*)&As[0][lrow * 32 + lhalf]     = a0;
        *(u32x4*)&As[0][lrow * 32 + lhalf + 8] = a1;
        *(u32x4*)&Bs[0][lrow * 32 + lhalf]     = b0;
        *(u32x4*)&Bs[0][lrow * 32 + lhalf + 8] = b1;
        __syncthreads();
        const ET* Asc = &As[0][0];
        const ET* Bsc = &Bs[0][0];
#endif
        V af[4], bf[2];
#pragma unroll
        for (int i = 0; i < 4; ++i) {
            FragU<V> f;
            const int r = (wm + i * 16 + lm) * 32;
            f.u[0] = *(const u32x4*)&Asc[r + lk8];        // VGPRs 0-3 (K half 0)
            f.u[1] = *(const u32x4*)&Asc[r + 16 + lk8];   // VGPRs 4-7 (K half 1)
            af[i] = f.v;
        }
#pragma unroll
        for (int j = 0; j < 2; ++j) {
            FragU<V> f;
            const int r = (wn + j * 16 + lm) * 32;
            f.u[0] = *(const u32x4*)&Bsc[r + lk8];
            f.u[1] = *(const u32x4*)&Bsc[r + 16 + lk8];
            bf[j] = f.v;
        }
#pragma unroll
        for (int i = 0; i < 4; ++i)
#pragma unroll
            for (int j = 0; j < 2; ++j)
                acc[i][j] = wmma_mma(af[i], bf[j], acc[i][j]);
#if USE_TDM
        if (wave == 0) __builtin_amdgcn_s_wait_tensorcnt(0);
        __syncthreads();
#endif
    }

    const int hi8 = (lane >> 4) * 8;   // C layout: VGPR r -> M = r + 8*(lane>=16)
    const int cn  = lane & 15;         //           N = lane & 15
#pragma unroll
    for (int i = 0; i < 4; ++i)
#pragma unroll
        for (int j = 0; j < 2; ++j)
#pragma unroll
            for (int r = 0; r < 8; ++r) {
                int m = gm0 + wm + i * 16 + hi8 + r;
                int n = gn0 + wn + j * 16 + cn;
                if (m < M && n < N) {
                    if (OT) C[(size_t)n * ldc + m] = acc[i][j][r];
                    else    C[(size_t)m * ldc + n] = acc[i][j][r];
                }
            }
}

// ---------------------------------------------------------------------------
// im2col with reflect-pad(1), fp32 -> bf16, dst[p][k], k = ci*9 + r*3 + s
// ---------------------------------------------------------------------------
__global__ void im2col_reflect(const float* __restrict__ src, bf16_t* __restrict__ dst,
                               int Cin, int H, int W, int stride, int Hout, int Wout)
{
    const int K = Cin * 9;
    const long long total = (long long)Hout * Wout * K;
    for (long long idx = (long long)blockIdx.x * 256 + threadIdx.x; idx < total;
         idx += (long long)gridDim.x * 256) {
        int k = (int)(idx % K);
        int p = (int)(idx / K);
        int ci = k / 9, rr = (k % 9) / 3, ss = k % 3;
        int oy = p / Wout, ox = p % Wout;
        int iy = oy * stride + rr - 1;
        int ix = ox * stride + ss - 1;
        iy = iy < 0 ? -iy : (iy >= H ? 2 * H - 2 - iy : iy);
        ix = ix < 0 ? -ix : (ix >= W ? 2 * W - 2 - ix : ix);
        dst[idx] = (bf16_t)src[(size_t)ci * H * W + (size_t)iy * W + ix];
    }
}

// ---------------------------------------------------------------------------
// InstanceNorm (+optional residual add, +optional PReLU). One block per channel.
// ---------------------------------------------------------------------------
__global__ __launch_bounds__(256) void inorm_prelu(
    const float* __restrict__ src, float* __restrict__ dst,
    const float* __restrict__ res, const float* __restrict__ alpha, int P)
{
    __shared__ float s1[256], s2[256];
    const int c = blockIdx.x;
    const float* x = src + (size_t)c * P;
    float sum = 0.f, sq = 0.f;
    for (int i = threadIdx.x; i < P; i += 256) { float v = x[i]; sum += v; sq += v * v; }
    s1[threadIdx.x] = sum; s2[threadIdx.x] = sq;
    __syncthreads();
    for (int o = 128; o > 0; o >>= 1) {
        if (threadIdx.x < o) { s1[threadIdx.x] += s1[threadIdx.x + o]; s2[threadIdx.x] += s2[threadIdx.x + o]; }
        __syncthreads();
    }
    const float mean = s1[0] / P;
    const float var  = s2[0] / P - mean * mean;
    const float rs   = rsqrtf(var + 1e-5f);
    const float a    = alpha ? alpha[0] : 0.f;
    float* y = dst + (size_t)c * P;
    const float* rz = res ? res + (size_t)c * P : nullptr;
    for (int i = threadIdx.x; i < P; i += 256) {
        float v = (x[i] - mean) * rs;
        if (rz) v += rz[i];
        if (alpha) v = v >= 0.f ? v : a * v;
        y[i] = v;
    }
}

// fp32 [C,P] -> ET [P,C] (transpose + convert)
template <typename ET>
__global__ void transpose_cvt(const float* __restrict__ src, ET* __restrict__ dst, int C, int P)
{
    const long long total = (long long)C * P;
    for (long long idx = (long long)blockIdx.x * 256 + threadIdx.x; idx < total;
         idx += (long long)gridDim.x * 256) {
        int c = (int)(idx % C);
        long long p = idx / C;
        dst[idx] = (ET)src[(size_t)c * P + p];
    }
}

template <typename ET>
__global__ void cvt_cast(const float* __restrict__ s, ET* __restrict__ d, long long n)
{
    for (long long i = (long long)blockIdx.x * 256 + threadIdx.x; i < n;
         i += (long long)gridDim.x * 256) d[i] = (ET)s[i];
}

// per-channel spatial mean of t[C,P] -> mean[C]
__global__ __launch_bounds__(256) void chan_mean(const float* __restrict__ t,
                                                 float* __restrict__ mean, int P)
{
    __shared__ float s[256];
    const float* x = t + (size_t)blockIdx.x * P;
    float sum = 0.f;
    for (int i = threadIdx.x; i < P; i += 256) sum += x[i];
    s[threadIdx.x] = sum; __syncthreads();
    for (int o = 128; o > 0; o >>= 1) {
        if (threadIdx.x < o) s[threadIdx.x] += s[threadIdx.x + o];
        __syncthreads();
    }
    if (threadIdx.x == 0) mean[blockIdx.x] = s[0] / P;
}

// center over spatial (mean[c]) then L2-normalize over channels; write [P,C] f16
__global__ void center_norm_T(const float* __restrict__ t, const float* __restrict__ mean,
                              f16_t* __restrict__ dst, int C, int P)
{
    const int p = blockIdx.x * 256 + threadIdx.x;
    if (p >= P) return;
    float acc = 0.f;
    for (int c = 0; c < C; ++c) { float v = t[(size_t)c * P + p] - mean[c]; acc += v * v; }
    const float inv = 1.0f / (sqrtf(acc) + 2.220446049250313e-16f);
    for (int c = 0; c < C; ++c) {
        float v = (t[(size_t)c * P + p] - mean[c]) * inv;
        dst[(size_t)p * C + c] = (f16_t)v;
    }
}

// row-wise: sim = max_n f ; fdiv = softmax(f / 0.005) in f16. One block per row.
__global__ __launch_bounds__(256) void softmax_row(const float* __restrict__ f,
                                                   f16_t* __restrict__ fdiv,
                                                   float* __restrict__ sim, int N)
{
    __shared__ float s[256];
    const float* fr = f + (size_t)blockIdx.x * N;
    float mx = -3.4e38f;
    for (int i = threadIdx.x; i < N; i += 256) mx = fmaxf(mx, fr[i]);
    s[threadIdx.x] = mx; __syncthreads();
    for (int o = 128; o > 0; o >>= 1) {
        if (threadIdx.x < o) s[threadIdx.x] = fmaxf(s[threadIdx.x], s[threadIdx.x + o]);
        __syncthreads();
    }
    mx = s[0];
    if (threadIdx.x == 0) sim[blockIdx.x] = mx;
    __syncthreads();
    float sum = 0.f;
    for (int i = threadIdx.x; i < N; i += 256) sum += __expf((fr[i] - mx) * 200.0f);
    s[threadIdx.x] = sum; __syncthreads();
    for (int o = 128; o > 0; o >>= 1) {
        if (threadIdx.x < o) s[threadIdx.x] += s[threadIdx.x + o];
        __syncthreads();
    }
    const float rs = 1.0f / s[0];
    f16_t* fd = fdiv + (size_t)blockIdx.x * N;
    for (int i = threadIdx.x; i < N; i += 256)
        fd[i] = (f16_t)(__expf((fr[i] - mx) * 200.0f) * rs);
}

// nearest 2x upsample: dst[c][y][x] = src[c][y/2][x/2]
__global__ void up2_k(const float* __restrict__ src, float* __restrict__ dst,
                      int C, int Hs, int Ws)
{
    const int Hd = Hs * 2, Wd = Ws * 2;
    const long long total = (long long)C * Hd * Wd;
    for (long long idx = (long long)blockIdx.x * 256 + threadIdx.x; idx < total;
         idx += (long long)gridDim.x * 256) {
        long long c = idx / ((long long)Hd * Wd);
        int rem = (int)(idx % ((long long)Hd * Wd));
        int y = rem / Wd, x = rem % Wd;
        dst[idx] = src[(size_t)c * Hs * Ws + (size_t)(y >> 1) * Ws + (x >> 1)];
    }
}

__global__ void avgpool_k(const float* __restrict__ src, float* __restrict__ dst,
                          int C, int Hs, int Ws, int k)
{
    const int Hd = Hs / k, Wd = Ws / k;
    const long long total = (long long)C * Hd * Wd;
    const float inv = 1.0f / (k * k);
    for (long long idx = (long long)blockIdx.x * 256 + threadIdx.x; idx < total;
         idx += (long long)gridDim.x * 256) {
        long long c = idx / ((long long)Hd * Wd);
        int rem = (int)(idx % ((long long)Hd * Wd));
        int y = rem / Wd, x = rem % Wd;
        float sum = 0.f;
        for (int dy = 0; dy < k; ++dy)
            for (int dx = 0; dx < k; ++dx)
                sum += src[(size_t)c * Hs * Ws + (size_t)(y * k + dy) * Ws + (x * k + dx)];
        dst[idx] = sum * inv;
    }
}

__global__ void copy_k(const float* __restrict__ s, float* __restrict__ d, long long n)
{
    for (long long i = (long long)blockIdx.x * 256 + threadIdx.x; i < n;
         i += (long long)gridDim.x * 256) d[i] = s[i];
}

// ---------------------------------------------------------------------------
// Host orchestration
// ---------------------------------------------------------------------------
extern "C" void kernel_launch(void* const* d_in, const int* in_sizes, int n_in,
                              void* d_out, int out_size, void* d_ws, size_t ws_size,
                              hipStream_t stream)
{
    (void)in_sizes; (void)n_in; (void)out_size; (void)ws_size;

    // ---- inputs (setup_inputs order; params flattened jax-style: sorted keys) ----
    const float* B_hist = (const float*)d_in[0];
    const float* A2 = (const float*)d_in[1];
    const float* A3 = (const float*)d_in[2];
    const float* A4 = (const float*)d_in[3];
    const float* A5 = (const float*)d_in[4];
    const float* Bi2 = (const float*)d_in[5];
    const float* Bi3 = (const float*)d_in[6];
    const float* Bi4 = (const float*)d_in[7];
    const float* Bi5 = (const float*)d_in[8];
    auto F = [&](int i) { return (const float*)d_in[i]; };
    // params: l2{c1b,c1w,c2b,c2w,p1,p2} l3{..} l4{..} l5{..} phi_b phi_w
    //         res[0..2]{a,c1b,c1w,c2b,c2w} theta_b theta_w
    const float* l2c1w = F(10); const float* l2c2w = F(12);
    const float* l2p1  = F(13); const float* l2p2  = F(14);
    const float* l3c1w = F(16); const float* l3c2w = F(18);
    const float* l3p1  = F(19); const float* l3p2  = F(20);
    const float* l4c1w = F(22); const float* l4c2w = F(24);
    const float* l4p1  = F(25); const float* l4p2  = F(26);
    const float* l5c1w = F(28); const float* l5c2w = F(30);
    const float* l5p1  = F(31); const float* l5p2  = F(32);
    const float* phi_w   = F(34);
    const float* res_a[3]   = { F(35), F(40), F(45) };
    const float* res_c1w[3] = { F(37), F(42), F(47) };
    const float* res_c2w[3] = { F(39), F(44), F(49) };
    const float* theta_w = F(51);
    // (all conv biases cancel under InstanceNorm / mean-centering -> skipped)

    // ---- workspace carving ----
    char* ws = (char*)d_ws;
    size_t off = 0;
    auto alloc = [&](size_t bytes) -> void* {
        off = (off + 255) & ~(size_t)255;
        void* p = ws + off;
        off += bytes;
        return p;
    };
    bf16_t* wt_l2c1 = (bf16_t*)alloc((size_t)73728 * 2);
    bf16_t* wt_l2c2 = (bf16_t*)alloc((size_t)73728 * 2);
    bf16_t* wt_l3c1 = (bf16_t*)alloc((size_t)147456 * 2);
    bf16_t* wt_l3c2 = (bf16_t*)alloc((size_t)73728 * 2);
    bf16_t* wt_l4c1 = (bf16_t*)alloc((size_t)589824 * 2);
    bf16_t* wt_l4c2 = (bf16_t*)alloc((size_t)147456 * 2);
    bf16_t* wt_l5c1 = (bf16_t*)alloc((size_t)1179648 * 2);
    bf16_t* wt_l5c2 = (bf16_t*)alloc((size_t)147456 * 2);
    bf16_t* wt_rc1[3], *wt_rc2[3];
    for (int r = 0; r < 3; ++r) {
        wt_rc1[r] = (bf16_t*)alloc((size_t)589824 * 2);
        wt_rc2[r] = (bf16_t*)alloc((size_t)589824 * 2);
    }
    bf16_t* wt_theta = (bf16_t*)alloc((size_t)65536 * 2);
    bf16_t* wt_phi   = (bf16_t*)alloc((size_t)65536 * 2);
    bf16_t* colbuf = (bf16_t*)alloc((size_t)9437184 * 2);           // max P*K
    float* stg0  = (float*)alloc((size_t)128 * 16384 * 4);          // 2M f32
    float* stg1  = (float*)alloc((size_t)256 * 4096 * 4);           // 1M f32
    float* feat  = (float*)alloc((size_t)256 * 4096 * 4);
    float* meanb = (float*)alloc(256 * 4);
    f16_t* thetaN = (f16_t*)alloc((size_t)2 * 4096 * 256 * 2);
    f16_t* phiN   = (f16_t*)alloc((size_t)2 * 4096 * 256 * 2);
    f16_t* bh16   = (f16_t*)alloc((size_t)2 * 512 * 4096 * 2);
    float* fmat   = (float*)alloc((size_t)4096 * 4096 * 4);         // per-batch
    f16_t* fdiv   = (f16_t*)alloc((size_t)4096 * 4096 * 2);         // per-batch
    float* simb   = (float*)alloc((size_t)2 * 4096 * 4);
    float* y1     = (float*)alloc((size_t)2 * 512 * 4096 * 4);

    auto NB = [](long long n) { return dim3((unsigned)((n + 255) / 256)); };

    // ---- weights fp32 -> bf16 ----
    auto CW = [&](const float* s, bf16_t* d, long long n) {
        cvt_cast<bf16_t><<<NB(n), 256, 0, stream>>>(s, d, n);
    };
    CW(l2c1w, wt_l2c1, 73728);  CW(l2c2w, wt_l2c2, 73728);
    CW(l3c1w, wt_l3c1, 147456); CW(l3c2w, wt_l3c2, 73728);
    CW(l4c1w, wt_l4c1, 589824); CW(l4c2w, wt_l4c2, 147456);
    CW(l5c1w, wt_l5c1, 1179648); CW(l5c2w, wt_l5c2, 147456);
    for (int r = 0; r < 3; ++r) { CW(res_c1w[r], wt_rc1[r], 589824); CW(res_c2w[r], wt_rc2[r], 589824); }
    CW(theta_w, wt_theta, 65536); CW(phi_w, wt_phi, 65536);

    // ---- launch helpers ----
    auto GEMM_BF = [&](const bf16_t* A, const bf16_t* B, float* C,
                       int M, int N, int K, int lda, int ldb, int ldc) {
        gemm_nt<bf16_t, false><<<dim3((N + 127) / 128, (M + 127) / 128), 256, 0, stream>>>(
            A, B, C, M, N, K, lda, ldb, ldc);
    };
    auto GEMM_F16 = [&](const f16_t* A, const f16_t* B, float* C,
                        int M, int N, int K, int lda, int ldb, int ldc, bool ot) {
        dim3 g((N + 127) / 128, (M + 127) / 128);
        if (ot) gemm_nt<f16_t, true ><<<g, 256, 0, stream>>>(A, B, C, M, N, K, lda, ldb, ldc);
        else    gemm_nt<f16_t, false><<<g, 256, 0, stream>>>(A, B, C, M, N, K, lda, ldb, ldc);
    };
    auto CONV = [&](const float* src, int Cin, int H, int W, int st,
                    const bf16_t* wt, int Cout, float* out) {
        int Ho = H / st, Wo = W / st;
        long long tot = (long long)Ho * Wo * Cin * 9;
        im2col_reflect<<<NB(tot), 256, 0, stream>>>(src, colbuf, Cin, H, W, st, Ho, Wo);
        GEMM_BF(wt, colbuf, out, Cout, Ho * Wo, Cin * 9, Cin * 9, Cin * 9, Ho * Wo);
    };
    auto INORM = [&](const float* s, float* d, const float* res, const float* a, int C, int P) {
        inorm_prelu<<<dim3(C), 256, 0, stream>>>(s, d, res, a, P);
    };
    auto UP2 = [&](const float* s, float* d, int C, int Hs, int Ws) {
        up2_k<<<NB((long long)C * Hs * Ws * 4), 256, 0, stream>>>(s, d, C, Hs, Ws);
    };

    // ---- pyramids: images 0,1 = A batch 0,1 -> theta ; 2,3 = B -> phi ----
    for (int img = 0; img < 4; ++img) {
        const int b = img & 1;
        const bool isA = img < 2;
        const float* x2 = (isA ? A2 : Bi2) + (size_t)b * 64 * 128 * 128;
        const float* x3 = (isA ? A3 : Bi3) + (size_t)b * 128 * 64 * 64;
        const float* x4 = (isA ? A4 : Bi4) + (size_t)b * 256 * 32 * 32;
        const float* x5 = (isA ? A5 : Bi5) + (size_t)b * 512 * 16 * 16;
        // l2: 64@128^2 -> 128@128^2 -> (s2) 64@64^2  => feat rows 0..63
        CONV(x2, 64, 128, 128, 1, wt_l2c1, 128, stg0);
        INORM(stg0, stg0, nullptr, l2p1, 128, 16384);
        CONV(stg0, 128, 128, 128, 2, wt_l2c2, 64, feat);
        INORM(feat, feat, nullptr, l2p2, 64, 4096);
        // l3 => feat rows 64..127
        CONV(x3, 128, 64, 64, 1, wt_l3c1, 128, stg0);
        INORM(stg0, stg0, nullptr, l3p1, 128, 4096);
        CONV(stg0, 128, 64, 64, 1, wt_l3c2, 64, feat + (size_t)64 * 4096);
        INORM(feat + (size_t)64 * 4096, feat + (size_t)64 * 4096, nullptr, l3p2, 64, 4096);
        // l4 => up2 -> feat rows 128..191
        CONV(x4, 256, 32, 32, 1, wt_l4c1, 256, stg0);
        INORM(stg0, stg0, nullptr, l4p1, 256, 1024);
        CONV(stg0, 256, 32, 32, 1, wt_l4c2, 64, stg1);
        INORM(stg1, stg1, nullptr, l4p2, 64, 1024);
        UP2(stg1, feat + (size_t)128 * 4096, 64, 32, 32);
        // l5 => cin -> up2 -> cin -> up2 -> feat rows 192..255
        CONV(x5, 512, 16, 16, 1, wt_l5c1, 256, stg0);
        INORM(stg0, stg0, nullptr, l5p1, 256, 256);
        UP2(stg0, stg1, 256, 16, 16);
        CONV(stg1, 256, 32, 32, 1, wt_l5c2, 64, stg0);
        INORM(stg0, stg0, nullptr, l5p2, 64, 1024);
        UP2(stg0, feat + (size_t)192 * 4096, 64, 32, 32);
        // 3 resblocks on feat [256,4096]
        for (int r = 0; r < 3; ++r) {
            CONV(feat, 256, 64, 64, 1, wt_rc1[r], 256, stg0);
            INORM(stg0, stg0, nullptr, res_a[r], 256, 4096);
            CONV(stg0, 256, 64, 64, 1, wt_rc2[r], 256, stg1);
            INORM(stg1, feat, feat, res_a[r], 256, 4096);   // IN + residual + PReLU
        }
        // theta/phi 1x1 conv -> center-norm -> [4096,256] f16
        transpose_cvt<bf16_t><<<NB((long long)4096 * 256), 256, 0, stream>>>(feat, colbuf, 256, 4096);
        GEMM_BF(isA ? wt_theta : wt_phi, colbuf, stg1, 256, 4096, 256, 256, 256, 4096);
        chan_mean<<<dim3(256), 256, 0, stream>>>(stg1, meanb, 4096);
        f16_t* dstN = (isA ? thetaN : phiN) + (size_t)b * 4096 * 256;
        center_norm_T<<<dim3(16), 256, 0, stream>>>(stg1, meanb, dstN, 256, 4096);
    }

    // ---- correlation + softmax + attention ----
    cvt_cast<f16_t><<<NB((long long)2 * 512 * 4096), 256, 0, stream>>>(
        B_hist, bh16, (long long)2 * 512 * 4096);
    for (int b = 0; b < 2; ++b) {
        GEMM_F16(thetaN + (size_t)b * 4096 * 256, phiN + (size_t)b * 4096 * 256, fmat,
                 4096, 4096, 256, 256, 256, 4096, false);
        softmax_row<<<dim3(4096), 256, 0, stream>>>(fmat, fdiv, simb + (size_t)b * 4096, 4096);
        GEMM_F16(fdiv, bh16 + (size_t)b * 512 * 4096, y1 + (size_t)b * 512 * 4096,
                 4096, 512, 4096, 4096, 4096, 4096, true);   // transposed store -> [512,4096]
    }

    // ---- outputs ----
    float* out = (float*)d_out;
    const size_t o0 = 0,          o1 = 16777216, o2 = 16809984, o3 = 21004288,
                 o4 = 21012480,   o5 = 22061056, o6 = 22063104, o7 = 22325248;
    UP2(y1,   out + o0, 1024, 64, 64);
    UP2(simb, out + o1, 2,    64, 64);
    copy_k<<<NB((long long)2 * 512 * 4096), 256, 0, stream>>>(y1, out + o2, (long long)2 * 512 * 4096);
    copy_k<<<NB(8192), 256, 0, stream>>>(simb, out + o3, 8192);
    avgpool_k<<<NB((long long)1024 * 32 * 32), 256, 0, stream>>>(y1,   out + o4, 1024, 64, 64, 2);
    avgpool_k<<<NB((long long)2 * 32 * 32),    256, 0, stream>>>(simb, out + o5, 2,    64, 64, 2);
    avgpool_k<<<NB((long long)1024 * 16 * 16), 256, 0, stream>>>(y1,   out + o6, 1024, 64, 64, 4);
    avgpool_k<<<NB((long long)2 * 16 * 16),    256, 0, stream>>>(simb, out + o7, 2,    64, 64, 4);
}
